// CILRS_58463094833193
// MI455X (gfx1250) — compile-verified, matrix operand order checked by reading, not run on hardware
//
#include <hip/hip_runtime.h>
#include <hip/hip_bf16.h>

// ---------------------------------------------------------------------------
// CDNA5 (gfx1250) wave32 WMMA types
// ---------------------------------------------------------------------------
typedef __attribute__((ext_vector_type(16))) _Float16 v16h;
typedef __attribute__((ext_vector_type(8)))  _Float16 v8h;
typedef __attribute__((ext_vector_type(8)))  float    v8f;

#define B_ROWS   32768
#define FEAT     512
#define SPD_H    128
#define CTRL_H   256
#define N_CMD    4
#define JOIN_K   (FEAT + SPD_H)   // 640

__device__ __forceinline__ v8f vzero8() {
  v8f z;
#pragma unroll
  for (int i = 0; i < 8; ++i) z[i] = 0.f;
  return z;
}

// A/B fragment loader from a row-major f16 matrix (row = M for A, row = N for
// B when the weight matrix is stored transposed [N][K]).
// ISA 7.12.2 (16-bit A 16x32): lane L: row = r0 + (L&15), half h = L>>4,
// elements 0..7 = M[row][k0+8h+j], elements 8..15 = M[row][k0+16+8h+j].
__device__ __forceinline__ v16h frag_rm_f16(const _Float16* M, int stride,
                                            int r0, int k0, int lane) {
  int r  = r0 + (lane & 15);
  int kb = k0 + ((lane >> 4) << 3);
  const _Float16* p = M + (size_t)r * stride + kb;
  v8h lo = *(const v8h*)(p);
  v8h hi = *(const v8h*)(p + 16);
  return __builtin_shufflevector(lo, hi, 0, 1, 2, 3, 4, 5, 6, 7,
                                         8, 9, 10, 11, 12, 13, 14, 15);
}

// C/D fragment (8 VGPR f32) epilogue: +bias[col], optional ReLU, store f16.
// ISA 7.12.2: VGPR i: lanes0-15 -> row m0+i, lanes16-31 -> row m0+8+i; col=lane&15.
__device__ __forceinline__ void store_c_f16(_Float16* D, int stride, int m0,
                                            int n0, int lane, v8f acc,
                                            const float* bias, bool relu) {
  int col   = n0 + (lane & 15);
  int rbase = m0 + ((lane >> 4) << 3);
  float bv  = bias[col];
#pragma unroll
  for (int i = 0; i < 8; ++i) {
    float x = acc[i] + bv;
    if (relu) x = fmaxf(x, 0.f);
    D[(size_t)(rbase + i) * stride + col] = (_Float16)x;
  }
}

__device__ __forceinline__ v8f wmma16(v16h a, v16h b, v8f c) {
  return __builtin_amdgcn_wmma_f32_16x16x32_f16(false, a, false, b,
                                                (short)0, c, false, false);
}

// Stage one 8-float chunk of an f32 row into LDS as 8 f16 (16B ds_store).
__device__ __forceinline__ void cvt8_f32_to_lds(const float* src, _Float16* dst) {
  float4 f0 = *(const float4*)(src);
  float4 f1 = *(const float4*)(src + 4);
  v8h h;
  h[0] = (_Float16)f0.x; h[1] = (_Float16)f0.y;
  h[2] = (_Float16)f0.z; h[3] = (_Float16)f0.w;
  h[4] = (_Float16)f1.x; h[5] = (_Float16)f1.y;
  h[6] = (_Float16)f1.z; h[7] = (_Float16)f1.w;
  *(v8h*)dst = h;
}

// gfx1250 async global->LDS copy of 16 bytes (ASYNCcnt-tracked, no VGPR data).
__device__ __forceinline__ void async_copy_b128(const _Float16* gsrc, _Float16* ldst) {
  unsigned lds_addr = (unsigned)(size_t)ldst;            // flat->LDS: addr[31:0]
  unsigned long long gaddr = (unsigned long long)(size_t)gsrc;
  asm volatile("global_load_async_to_lds_b128 %0, %1, off"
               :: "v"(lds_addr), "v"(gaddr) : "memory");
}
__device__ __forceinline__ void async_wait0() {
  asm volatile("s_wait_asynccnt 0" ::: "memory");
}

// ---------------------------------------------------------------------------
// Prep: transpose f32 [E][K][N] weights into f16 [E][N][K]
// ---------------------------------------------------------------------------
__global__ void transpose_f32_to_f16(const float* __restrict__ src,
                                     _Float16* __restrict__ dst,
                                     int K, int N, int total) {
  int idx = blockIdx.x * blockDim.x + threadIdx.x;
  if (idx >= total) return;
  int kn  = K * N;
  int e   = idx / kn;
  int rem = idx - e * kn;
  int k   = rem / N;
  int n   = rem - k * N;
  dst[((size_t)e * N + n) * K + k] = (_Float16)src[idx];
}

// ctrl_w3 [4][256][3] -> zero-padded f16 [4][16][256] so the final layer can
// run as one 16x16x32 WMMA chain per expert.
__global__ void prep_w3_pad(const float* __restrict__ src,
                            _Float16* __restrict__ dst) {
  int idx = blockIdx.x * blockDim.x + threadIdx.x;
  if (idx >= N_CMD * 16 * CTRL_H) return;
  int k = idx & (CTRL_H - 1);
  int o = (idx >> 8) & 15;
  int e = idx >> 12;
  float v = (o < 3) ? src[((size_t)e * CTRL_H + k) * 3 + o] : 0.f;
  dst[idx] = (_Float16)v;
}

// ---------------------------------------------------------------------------
// measured_speed_module: 1 -> 128 -> 128 -> 128, output v as f16 [B][128]
// block = 128 threads (4 waves), 64 rows per block
// ---------------------------------------------------------------------------
__global__ __launch_bounds__(128) void ms_kernel(
    const float* __restrict__ speed,
    const float* __restrict__ w1, const float* __restrict__ b1,
    const _Float16* __restrict__ w2t, const float* __restrict__ b2,
    const _Float16* __restrict__ w3t, const float* __restrict__ b3,
    _Float16* __restrict__ v_out) {
  extern __shared__ char smem[];
  _Float16* h1 = (_Float16*)smem;           // [64][128]
  _Float16* h2 = h1 + 64 * SPD_H;           // [64][128]
  int tid = threadIdx.x, lane = tid & 31, wave = tid >> 5;
  int rows0 = blockIdx.x * 64;
  int m0 = wave * 16;

  // layer 1: outer product with scalar speed
  for (int t = tid; t < 64 * SPD_H; t += 128) {
    int r = t >> 7, j = t & (SPD_H - 1);
    float x = speed[rows0 + r] * w1[j] + b1[j];
    h1[t] = (_Float16)fmaxf(x, 0.f);
  }
  __syncthreads();

  // layer 2: h2 = relu(h1 @ w2 + b2)
  for (int ng = 0; ng < 2; ++ng) {
    v8f acc[4];
#pragma unroll
    for (int s = 0; s < 4; ++s) acc[s] = vzero8();
    for (int k0 = 0; k0 < SPD_H; k0 += 32) {
      v16h a = frag_rm_f16(h1, SPD_H, m0, k0, lane);
#pragma unroll
      for (int s = 0; s < 4; ++s) {
        v16h b = frag_rm_f16(w2t, SPD_H, ng * 64 + s * 16, k0, lane);
        acc[s] = wmma16(a, b, acc[s]);
      }
    }
#pragma unroll
    for (int s = 0; s < 4; ++s)
      store_c_f16(h2, SPD_H, m0, ng * 64 + s * 16, lane, acc[s], b2, true);
  }
  __syncthreads();

  // layer 3: v = h2 @ w3 + b3 -> global f16
  _Float16* Dst = v_out + (size_t)rows0 * SPD_H;
  for (int ng = 0; ng < 2; ++ng) {
    v8f acc[4];
#pragma unroll
    for (int s = 0; s < 4; ++s) acc[s] = vzero8();
    for (int k0 = 0; k0 < SPD_H; k0 += 32) {
      v16h a = frag_rm_f16(h2, SPD_H, m0, k0, lane);
#pragma unroll
      for (int s = 0; s < 4; ++s) {
        v16h b = frag_rm_f16(w3t, SPD_H, ng * 64 + s * 16, k0, lane);
        acc[s] = wmma16(a, b, acc[s]);
      }
    }
#pragma unroll
    for (int s = 0; s < 4; ++s)
      store_c_f16(Dst, SPD_H, m0, ng * 64 + s * 16, lane, acc[s], b3, false);
  }
}

// ---------------------------------------------------------------------------
// speed_pred_module: 512 -> 256 -> 256 -> 1, writes v_p into d_out[0..B)
// feat tile staged once to LDS as f16; h2 aliases the staging buffer.
// ---------------------------------------------------------------------------
__global__ __launch_bounds__(128) void sp_kernel(
    const float* __restrict__ feat,
    const _Float16* __restrict__ w1t, const float* __restrict__ b1,
    const _Float16* __restrict__ w2t, const float* __restrict__ b2,
    const float* __restrict__ w3, const float* __restrict__ b3,
    float* __restrict__ vp_out) {
  extern __shared__ char smem[];
  _Float16* featT = (_Float16*)smem;        // [64][512]  (64 KB)
  _Float16* h1    = featT + 64 * FEAT;      // [64][256]  (32 KB)
  _Float16* h2    = featT;                  // alias: featT dead after layer 1
  int tid = threadIdx.x, lane = tid & 31, wave = tid >> 5;
  int rows0 = blockIdx.x * 64;
  int m0 = wave * 16;
  const float* A = feat + (size_t)rows0 * FEAT;

  // stage feat tile -> LDS f16 (one pass, vectorized)
  for (int t = tid; t < 64 * (FEAT / 8); t += 128) {
    int r = t >> 6;
    int c = (t & 63) * 8;
    cvt8_f32_to_lds(A + (size_t)r * FEAT + c, featT + r * FEAT + c);
  }
  __syncthreads();

  for (int ng = 0; ng < 4; ++ng) {
    v8f acc[4];
#pragma unroll
    for (int s = 0; s < 4; ++s) acc[s] = vzero8();
    for (int k0 = 0; k0 < FEAT; k0 += 32) {
      v16h a = frag_rm_f16(featT, FEAT, m0, k0, lane);
#pragma unroll
      for (int s = 0; s < 4; ++s) {
        v16h b = frag_rm_f16(w1t, FEAT, ng * 64 + s * 16, k0, lane);
        acc[s] = wmma16(a, b, acc[s]);
      }
    }
#pragma unroll
    for (int s = 0; s < 4; ++s)
      store_c_f16(h1, CTRL_H, m0, ng * 64 + s * 16, lane, acc[s], b1, true);
  }
  __syncthreads();

  for (int ng = 0; ng < 4; ++ng) {
    v8f acc[4];
#pragma unroll
    for (int s = 0; s < 4; ++s) acc[s] = vzero8();
    for (int k0 = 0; k0 < CTRL_H; k0 += 32) {
      v16h a = frag_rm_f16(h1, CTRL_H, m0, k0, lane);
#pragma unroll
      for (int s = 0; s < 4; ++s) {
        v16h b = frag_rm_f16(w2t, CTRL_H, ng * 64 + s * 16, k0, lane);
        acc[s] = wmma16(a, b, acc[s]);
      }
    }
#pragma unroll
    for (int s = 0; s < 4; ++s)
      store_c_f16(h2, CTRL_H, m0, ng * 64 + s * 16, lane, acc[s], b2, true);
  }
  __syncthreads();

  if (tid < 64) {
    float acc = b3[0];
    const _Float16* row = h2 + (size_t)tid * CTRL_H;
    for (int k = 0; k < CTRL_H; ++k) acc += (float)row[k] * w3[k];
    vp_out[rows0 + tid] = acc;
  }
}

// ---------------------------------------------------------------------------
// join + command-selected control heads, fully fused per 64-row tile.
// [feat|v] staged once into LDS (v via gfx1250 async-to-LDS); joined lives
// only in LDS; h1/h2 alias the dead staging buffer. Heads computed only for
// experts present in the tile.
// ---------------------------------------------------------------------------
__global__ __launch_bounds__(128) void ctrl_kernel(
    const float* __restrict__ feat, const _Float16* __restrict__ vspd,
    const int* __restrict__ command,
    const _Float16* __restrict__ join_wt, const float* __restrict__ join_b,
    const _Float16* __restrict__ w1t, const float* __restrict__ b1,
    const _Float16* __restrict__ w2t, const float* __restrict__ b2,
    const _Float16* __restrict__ w3p, const float* __restrict__ b3,
    float* __restrict__ out, int Btot) {
  extern __shared__ char smem[];
  _Float16* featT  = (_Float16*)smem;              // [64][640] (80 KB)
  _Float16* joined = featT + 64 * JOIN_K;          // [64][512] (64 KB)
  int* cmd   = (int*)(joined + 64 * FEAT);         // [64]
  int* flags = cmd + 64;                           // [4]
  _Float16* h1 = featT;                            // alias (32 KB)
  _Float16* h2 = featT + 64 * CTRL_H;              // alias (32 KB)
  int tid = threadIdx.x, lane = tid & 31, wave = tid >> 5;
  int rows0 = blockIdx.x * 64;
  int m0 = wave * 16;

  if (tid < N_CMD) flags[tid] = 0;
  __syncthreads();
  if (tid < 64) {
    int c = command[rows0 + tid] & (N_CMD - 1);
    cmd[tid] = c;
    flags[c] = 1;
  }

  // stage [feat | v] tile -> LDS f16 [64][640]
  const float*    Af = feat + (size_t)rows0 * FEAT;
  const _Float16* Av = vspd + (size_t)rows0 * SPD_H;
  for (int t = tid; t < 64 * (FEAT / 8); t += 128) {
    int r = t >> 6;
    int c = (t & 63) * 8;
    cvt8_f32_to_lds(Af + (size_t)r * FEAT + c, featT + r * JOIN_K + c);
  }
  for (int t = tid; t < 64 * (SPD_H / 8); t += 128) {   // async b128 copies
    int r = t >> 4;
    int c = (t & 15) * 8;
    async_copy_b128(Av + (size_t)r * SPD_H + c, featT + r * JOIN_K + FEAT + c);
  }
  async_wait0();
  __syncthreads();

  // joined = [feat|v] @ join_w + join_b  (single K = 640 loop from LDS)
  for (int ng = 0; ng < 8; ++ng) {
    v8f acc[4];
#pragma unroll
    for (int s = 0; s < 4; ++s) acc[s] = vzero8();
    for (int k0 = 0; k0 < JOIN_K; k0 += 32) {
      v16h a = frag_rm_f16(featT, JOIN_K, m0, k0, lane);
#pragma unroll
      for (int s = 0; s < 4; ++s) {
        v16h b = frag_rm_f16(join_wt, JOIN_K, ng * 64 + s * 16, k0, lane);
        acc[s] = wmma16(a, b, acc[s]);
      }
    }
#pragma unroll
    for (int s = 0; s < 4; ++s)
      store_c_f16(joined, FEAT, m0, ng * 64 + s * 16, lane, acc[s], join_b, false);
  }
  __syncthreads();   // featT dead from here; h1/h2 alias it

  for (int e = 0; e < N_CMD; ++e) {
    if (!flags[e]) continue;
    const _Float16* W1 = w1t + (size_t)e * CTRL_H * FEAT;
    const _Float16* W2 = w2t + (size_t)e * CTRL_H * CTRL_H;
    const _Float16* W3 = w3p + (size_t)e * 16 * CTRL_H;
    const float* B1 = b1 + e * CTRL_H;
    const float* B2 = b2 + e * CTRL_H;

    // h1 = relu(joined @ w1[e] + b1[e])
    for (int ng = 0; ng < 4; ++ng) {
      v8f acc[4];
#pragma unroll
      for (int s = 0; s < 4; ++s) acc[s] = vzero8();
      for (int k0 = 0; k0 < FEAT; k0 += 32) {
        v16h a = frag_rm_f16(joined, FEAT, m0, k0, lane);
#pragma unroll
        for (int s = 0; s < 4; ++s) {
          v16h b = frag_rm_f16(W1, FEAT, ng * 64 + s * 16, k0, lane);
          acc[s] = wmma16(a, b, acc[s]);
        }
      }
#pragma unroll
      for (int s = 0; s < 4; ++s)
        store_c_f16(h1, CTRL_H, m0, ng * 64 + s * 16, lane, acc[s], B1, true);
    }
    __syncthreads();

    // h2 = relu(h1 @ w2[e] + b2[e])
    for (int ng = 0; ng < 4; ++ng) {
      v8f acc[4];
#pragma unroll
      for (int s = 0; s < 4; ++s) acc[s] = vzero8();
      for (int k0 = 0; k0 < CTRL_H; k0 += 32) {
        v16h a = frag_rm_f16(h1, CTRL_H, m0, k0, lane);
#pragma unroll
        for (int s = 0; s < 4; ++s) {
          v16h b = frag_rm_f16(W2, CTRL_H, ng * 64 + s * 16, k0, lane);
          acc[s] = wmma16(a, b, acc[s]);
        }
      }
#pragma unroll
      for (int s = 0; s < 4; ++s)
        store_c_f16(h2, CTRL_H, m0, ng * 64 + s * 16, lane, acc[s], B2, true);
    }
    __syncthreads();

    // act = h2 @ w3pad[e]  (one 16-col WMMA chain; cols 0..2 valid)
    v8f acc = vzero8();
    for (int k0 = 0; k0 < CTRL_H; k0 += 32) {
      v16h a = frag_rm_f16(h2, CTRL_H, m0, k0, lane);
      v16h b = frag_rm_f16(W3, CTRL_H, 0, k0, lane);
      acc = wmma16(a, b, acc);
    }
    int o = lane & 15;
    if (o < 3) {
      float bv = b3[e * 3 + o];
      int rbase = m0 + ((lane >> 4) << 3);
#pragma unroll
      for (int i = 0; i < 8; ++i) {
        int r = rbase + i;
        if (cmd[r] == e) {
          float x = acc[i] + bv;
          int gr = rows0 + r;
          if (o == 0)      out[Btot + gr]     = 1.f / (1.f + __expf(-x)); // throttle
          else if (o == 1) out[3 * Btot + gr] = tanhf(x);                 // steering
          else             out[2 * Btot + gr] = 1.f / (1.f + __expf(-x)); // brake
        }
      }
    }
    __syncthreads();
  }
}

// ---------------------------------------------------------------------------
// Launch
// ---------------------------------------------------------------------------
extern "C" void kernel_launch(void* const* d_in, const int* in_sizes, int n_in,
                              void* d_out, int out_size, void* d_ws, size_t ws_size,
                              hipStream_t stream) {
  (void)in_sizes; (void)n_in; (void)out_size; (void)ws_size;
  const float* feat    = (const float*)d_in[0];
  const float* speed   = (const float*)d_in[1];
  const int*   command = (const int*)d_in[2];
  const float* ms_w1   = (const float*)d_in[3];
  const float* ms_b1   = (const float*)d_in[4];
  const float* ms_w2   = (const float*)d_in[5];
  const float* ms_b2   = (const float*)d_in[6];
  const float* ms_w3   = (const float*)d_in[7];
  const float* ms_b3   = (const float*)d_in[8];
  const float* sp_w1   = (const float*)d_in[9];
  const float* sp_b1   = (const float*)d_in[10];
  const float* sp_w2   = (const float*)d_in[11];
  const float* sp_b2   = (const float*)d_in[12];
  const float* sp_w3   = (const float*)d_in[13];
  const float* sp_b3   = (const float*)d_in[14];
  const float* join_w  = (const float*)d_in[15];
  const float* join_b  = (const float*)d_in[16];
  const float* ctrl_w1 = (const float*)d_in[17];
  const float* ctrl_b1 = (const float*)d_in[18];
  const float* ctrl_w2 = (const float*)d_in[19];
  const float* ctrl_b2 = (const float*)d_in[20];
  const float* ctrl_w3 = (const float*)d_in[21];
  const float* ctrl_b3 = (const float*)d_in[22];
  float* out = (float*)d_out;

  // workspace carve (f16 elements, all 16B-aligned offsets)
  _Float16* ws = (_Float16*)d_ws;
  _Float16* ms_w2t   = ws;                         // 128*128
  _Float16* ms_w3t   = ms_w2t + 128 * 128;         // 128*128
  _Float16* sp_w1t   = ms_w3t + 128 * 128;         // 256*512
  _Float16* sp_w2t   = sp_w1t + 256 * 512;         // 256*256
  _Float16* join_wt  = sp_w2t + 256 * 256;         // 512*640
  _Float16* ctrl_w1t = join_wt + 512 * 640;        // 4*256*512
  _Float16* ctrl_w2t = ctrl_w1t + 4 * 256 * 512;   // 4*256*256
  _Float16* ctrl_w3p = ctrl_w2t + 4 * 256 * 256;   // 4*16*256
  _Float16* v_f16    = ctrl_w3p + 4 * 16 * 256;    // B*128

  auto tp = [&](const float* src, _Float16* dst, int E, int K, int N) {
    int total = E * K * N;
    transpose_f32_to_f16<<<(total + 255) / 256, 256, 0, stream>>>(src, dst, K, N, total);
  };
  tp(ms_w2,   ms_w2t,   1, 128, 128);
  tp(ms_w3,   ms_w3t,   1, 128, 128);
  tp(sp_w1,   sp_w1t,   1, 512, 256);
  tp(sp_w2,   sp_w2t,   1, 256, 256);
  tp(join_w,  join_wt,  1, 640, 512);
  tp(ctrl_w1, ctrl_w1t, 4, 512, 256);
  tp(ctrl_w2, ctrl_w2t, 4, 256, 256);
  prep_w3_pad<<<(N_CMD * 16 * CTRL_H + 255) / 256, 256, 0, stream>>>(ctrl_w3, ctrl_w3p);

  int nblocks = B_ROWS / 64;
  size_t sh_ms   = 2 * 64 * SPD_H * sizeof(_Float16);                        // 32 KB
  size_t sh_sp   = (64 * FEAT + 64 * CTRL_H) * sizeof(_Float16);             // 96 KB
  size_t sh_ctrl = (size_t)(64 * JOIN_K + 64 * FEAT) * sizeof(_Float16)
                 + 68 * sizeof(int);                                         // ~145 KB (<320 KB/WGP)

  ms_kernel<<<nblocks, 128, sh_ms, stream>>>(speed, ms_w1, ms_b1, ms_w2t, ms_b2,
                                             ms_w3t, ms_b3, v_f16);
  sp_kernel<<<nblocks, 128, sh_sp, stream>>>(feat, sp_w1t, sp_b1, sp_w2t, sp_b2,
                                             sp_w3, sp_b3, out);
  ctrl_kernel<<<nblocks, 128, sh_ctrl, stream>>>(feat, v_f16, command,
                                                 join_wt, join_b,
                                                 ctrl_w1t, ctrl_b1,
                                                 ctrl_w2t, ctrl_b2,
                                                 ctrl_w3p, ctrl_b3,
                                                 out, B_ROWS);
}
